// LGHGCLNet_56057913147947
// MI455X (gfx1250) — matrix operand order repaired and checked by reading, not compiled, not run on hardware
//
#include <hip/hip_runtime.h>
#include <hip/hip_bf16.h>

// ---------------- problem constants (match reference) ----------------
#define NNODES 100000
#define NEDGES 500000
#define RREL   3
#define HDIM   256
#define BNEPS  1e-5f

typedef __attribute__((ext_vector_type(16))) __bf16 v16bf;
typedef __attribute__((ext_vector_type(8)))  __bf16 v8bf;
typedef __attribute__((ext_vector_type(8)))  float  v8f;

union V16U { v16bf v; v8bf h[2]; };

// ---------------- small utility kernels ----------------
__global__ void zero_f32_kernel(float* p, long long n) {
    long long i = (long long)blockIdx.x * blockDim.x + threadIdx.x;
    if (i < n) p[i] = 0.f;
}

__global__ void init_out_kernel(float* o, const float* ob, int n) {
    int i = blockIdx.x * blockDim.x + threadIdx.x;
    if (i < n) o[i] = ob[0];
}

// Transpose + convert weights:  d (256 x KT, bf16)  <-  [s0 (K0 x 256) ; s1 ((KT-K0) x 256)] fp32
__global__ void transpose_w_kernel(const float* __restrict__ s0, const float* __restrict__ s1,
                                   __bf16* __restrict__ d, int K0, int KT) {
    long long i = (long long)blockIdx.x * blockDim.x + threadIdx.x;
    long long total = (long long)HDIM * KT;
    if (i < total) {
        int n = (int)(i / KT);
        int k = (int)(i % KT);
        float v = (k < K0) ? s0[(size_t)k * HDIM + n] : s1[(size_t)(k - K0) * HDIM + n];
        d[i] = (__bf16)v;
    }
}

// ---------------- graph kernels ----------------
__global__ void count_kernel(const int* __restrict__ ei, const int* __restrict__ et,
                             float* __restrict__ cntRow, float* __restrict__ cntCol, int E) {
    int e = blockIdx.x * blockDim.x + threadIdx.x;
    if (e < E) {
        int r = ei[e], c = ei[E + e], t = et[e];
        atomicAdd(&cntRow[r * RREL + t], 1.f);
        atomicAdd(&cntCol[c * RREL + t], 1.f);
    }
}

__global__ void invert_scre_kernel(float* cnt, float* relinv, int n) {
    int i = blockIdx.x * blockDim.x + threadIdx.x;
    if (i < n) {
        float s = 0.f;
#pragma unroll
        for (int r = 0; r < RREL; ++r) {
            float c = cnt[i * RREL + r];
            s += (c > 0.f) ? 1.f : 0.f;
            cnt[i * RREL + r] = (c > 0.f) ? (1.f / c) : 0.f;
        }
        relinv[i] = 1.f / fmaxf(s, 1.f);
    }
}

__global__ void invert_flat_kernel(float* cnt, int n) {
    int i = blockIdx.x * blockDim.x + threadIdx.x;
    if (i < n) { float c = cnt[i]; cnt[i] = (c > 0.f) ? (1.f / c) : 0.f; }
}

// SCRE: tmp[row] += x[col] * invcnt[row, etype]  (one wave per edge, vectorized fp32 gather)
__global__ __launch_bounds__(256)
void scre_scatter_kernel(const int* __restrict__ ei, const int* __restrict__ et,
                         const float* __restrict__ x, const float* __restrict__ invcnt,
                         float* __restrict__ tmp, int E) {
    int wid  = (int)((blockIdx.x * (long long)blockDim.x + threadIdx.x) >> 5);
    int lane = threadIdx.x & 31;
    if (wid >= E) return;
    int r = ei[wid], c = ei[E + wid], t = et[wid];
    float w = invcnt[r * RREL + t];
    const float4* src = reinterpret_cast<const float4*>(&x[(size_t)c * HDIM + lane * 8]);
    float4 v0 = src[0], v1 = src[1];
    float* dst = &tmp[(size_t)r * HDIM + lane * 8];
    atomicAdd(dst + 0, v0.x * w); atomicAdd(dst + 1, v0.y * w);
    atomicAdd(dst + 2, v0.z * w); atomicAdd(dst + 3, v0.w * w);
    atomicAdd(dst + 4, v1.x * w); atomicAdd(dst + 5, v1.y * w);
    atomicAdd(dst + 6, v1.z * w); atomicAdd(dst + 7, v1.w * w);
}

// h = x - tmp * relinv, stored bf16
__global__ void compute_h_kernel(const float* __restrict__ x, const float* __restrict__ tmp,
                                 const float* __restrict__ relinv, __bf16* __restrict__ h,
                                 long long total) {
    long long i = (long long)blockIdx.x * blockDim.x + threadIdx.x;
    if (i < total) {
        long long n = i >> 8;
        h[i] = (__bf16)(x[i] - tmp[i] * relinv[n]);
    }
}

// RGCN aggregation: agg[col, etype, :] += feat[row,:] * invcnt[col, etype]
__global__ __launch_bounds__(256)
void rgcn_scatter_kernel(const int* __restrict__ ei, const int* __restrict__ et,
                         const __bf16* __restrict__ feat, const float* __restrict__ invcnt,
                         float* __restrict__ agg, int E) {
    int wid  = (int)((blockIdx.x * (long long)blockDim.x + threadIdx.x) >> 5);
    int lane = threadIdx.x & 31;
    if (wid >= E) return;
    int r = ei[wid], c = ei[E + wid], t = et[wid];
    float w = invcnt[c * RREL + t];
    v8bf hv = *reinterpret_cast<const v8bf*>(&feat[(size_t)r * HDIM + lane * 8]);
    float* dst = &agg[(size_t)c * (RREL * HDIM) + t * HDIM + lane * 8];
#pragma unroll
    for (int i = 0; i < 8; ++i) atomicAdd(dst + i, (float)hv[i] * w);
}

// ---------------- WMMA GEMM with fused bias + BN + ReLU epilogue ----------------
// Logical A = [A0 (TA0, K0 cols) | A1 (TA1, K1 cols)], row-major.
// Wt = transposed bf16 weights: 256 rows (out col n) x KT cols (k), row-major.
// B tiles staged with double-buffered GLOBAL_LOAD_ASYNC_TO_LDS_B128 (ASYNCcnt),
// prefetching tile i+1 while WMMAs consume tile i.
// OMODE 0: store relu(bn(..)) as bf16 (nRows x 256)
// OMODE 1: head fusion: atomicAdd(out[row], relu(bn(..)) * headw[col])

__device__ __forceinline__ v8bf loadRun8(const __bf16* p) {
    return *reinterpret_cast<const v8bf*>(p);
}
__device__ __forceinline__ v8bf loadRun8(const float* p) {
    float4 f0 = *reinterpret_cast<const float4*>(p);
    float4 f1 = *reinterpret_cast<const float4*>(p + 4);
    v8bf r;
    r[0] = (__bf16)f0.x; r[1] = (__bf16)f0.y; r[2] = (__bf16)f0.z; r[3] = (__bf16)f0.w;
    r[4] = (__bf16)f1.x; r[5] = (__bf16)f1.y; r[6] = (__bf16)f1.z; r[7] = (__bf16)f1.w;
    return r;
}

template<typename TA0, int K0, typename TA1, int K1>
__device__ __forceinline__ v8bf fetchRun(const TA0* A0, const TA1* A1, int row, int k) {
    // k is a multiple of 8; an 8-run never crosses the K0 boundary (K0 % 32 == 0)
    if (K1 == 0 || k < K0) return loadRun8(&A0[(size_t)row * K0 + k]);
    return loadRun8(&A1[(size_t)row * K1 + (k - K0)]);
}

template<typename TA0, int K0, typename TA1, int K1, int OMODE>
__global__ __launch_bounds__(256)
void gemm_bn_relu_kernel(const TA0* __restrict__ A0, const TA1* __restrict__ A1,
                         const __bf16* __restrict__ Wt,
                         const float* __restrict__ bias,
                         const float* __restrict__ bn_g, const float* __restrict__ bn_b,
                         const float* __restrict__ bn_m, const float* __restrict__ bn_v,
                         void* __restrict__ outp, const float* __restrict__ headw,
                         int nRows) {
    constexpr int KT = K0 + K1;
    constexpr int NT = KT / 32;                      // number of 32-wide k tiles
    constexpr int BSTR = 40;                         // padded row stride (bf16), 80B: 16B-aligned
    __shared__ __align__(16) __bf16 BsT[2][64 * BSTR]; // double-buffered 64col x 32k tiles (10KB)

    const int tid  = threadIdx.x;
    const int wave = tid >> 5;
    const int lane = tid & 31;
    const int half = lane >> 4;
    const int m16  = lane & 15;
    const int rowBase = blockIdx.x * 128 + wave * 16;  // wave: 16 rows x 64 cols
    const int cb      = blockIdx.y * 64;
    // clamp OOB rows: their accumulators are garbage but never stored (epilogue guard)
    const int arow = min(rowBase + m16, nRows - 1);

    // staging coordinates: thread owns 8 contiguous k of one output-column (16B)
    const int sn = tid >> 2;          // 0..63
    const int sk = (tid & 3) * 8;     // 0,8,16,24
    const __bf16* wbase = &Wt[(size_t)(cb + sn) * KT + sk];

    auto issueTile = [&](int t, int buf) {
        unsigned lds = (unsigned)(uintptr_t)&BsT[buf][sn * BSTR + sk];
        unsigned long long ga = (unsigned long long)(uintptr_t)(wbase + t * 32);
        asm volatile("global_load_async_to_lds_b128 %0, %1, off"
                     :: "v"(lds), "v"(ga) : "memory");
    };

    v8f acc[4] = { {0,0,0,0,0,0,0,0}, {0,0,0,0,0,0,0,0},
                   {0,0,0,0,0,0,0,0}, {0,0,0,0,0,0,0,0} };

    issueTile(0, 0);                                  // prologue prefetch
    for (int i = 0; i < NT; ++i) {
        const int k0 = i * 32;
        const int buf = i & 1;
        asm volatile("s_wait_asynccnt 0x0" ::: "memory");  // my part of tile i is in LDS
        __syncthreads();                                    // everyone's part is in LDS
        if (i + 1 < NT) issueTile(i + 1, buf ^ 1);          // prefetch overlaps the WMMAs

        // A fragment: two contiguous 8-runs (ISA 16-bit A 16x32 lane layout)
        V16U a;
        a.h[0] = fetchRun<TA0, K0, TA1, K1>(A0, A1, arow, k0 + 8 * half);
        a.h[1] = fetchRun<TA0, K0, TA1, K1>(A0, A1, arow, k0 + 16 + 8 * half);

#pragma unroll
        for (int ct = 0; ct < 4; ++ct) {
            // B fragment: column n = ct*16+m16, k = 8*half .. +8 and 16+8*half .. +8 -> 2x ds_load_b128
            const __bf16* bp = &BsT[buf][(ct * 16 + m16) * BSTR + 8 * half];
            V16U b;
            b.h[0] = *reinterpret_cast<const v8bf*>(bp);
            b.h[1] = *reinterpret_cast<const v8bf*>(bp + 16);
            acc[ct] = __builtin_amdgcn_wmma_f32_16x16x32_bf16(
                false, a.v, false, b.v, (short)0, acc[ct], false, false);
        }
    }

    // epilogue: y = (acc + bias - m) * g/sqrt(v+eps) + b, relu
#pragma unroll
    for (int ct = 0; ct < 4; ++ct) {
        int j = cb + ct * 16 + m16;
        float sc = bn_g[j] * rsqrtf(bn_v[j] + BNEPS);
        float sh = (bias[j] - bn_m[j]) * sc + bn_b[j];
#pragma unroll
        for (int p = 0; p < 8; ++p) {
            int row = rowBase + p + 8 * half;   // C/D layout: VGPR p -> M = p + 8*half
            if (row < nRows) {
                float y = fmaxf(acc[ct][p] * sc + sh, 0.f);
                if (OMODE == 0) {
                    reinterpret_cast<__bf16*>(outp)[(size_t)row * HDIM + j] = (__bf16)y;
                } else {
                    atomicAdd(reinterpret_cast<float*>(outp) + row, y * headw[j]);
                }
            }
        }
    }
}

// ---------------- launcher ----------------
extern "C" void kernel_launch(void* const* d_in, const int* in_sizes, int n_in,
                              void* d_out, int out_size, void* d_ws, size_t ws_size,
                              hipStream_t stream) {
    (void)in_sizes; (void)n_in; (void)ws_size; (void)out_size;

    const float* x       = (const float*)d_in[0];
    const int*   ei      = (const int*)d_in[1];      // (2,E): row = ei[e], col = ei[E+e]
    const int*   et      = (const int*)d_in[2];
    const float* mlp_w1  = (const float*)d_in[3];
    const float* mlp_b1  = (const float*)d_in[4];
    const float* mbn1_g  = (const float*)d_in[5];
    const float* mbn1_b  = (const float*)d_in[6];
    const float* mbn1_m  = (const float*)d_in[7];
    const float* mbn1_v  = (const float*)d_in[8];
    const float* mlp_w2  = (const float*)d_in[9];
    const float* mlp_b2  = (const float*)d_in[10];
    const float* mbn2_g  = (const float*)d_in[11];
    const float* mbn2_b  = (const float*)d_in[12];
    const float* mbn2_m  = (const float*)d_in[13];
    const float* mbn2_v  = (const float*)d_in[14];
    const float* r1_wrel = (const float*)d_in[15];   // (3,256,256) contiguous => rows 0..767
    const float* r1_wroot= (const float*)d_in[16];
    const float* r1_b    = (const float*)d_in[17];
    const float* gbn1_g  = (const float*)d_in[18];
    const float* gbn1_b  = (const float*)d_in[19];
    const float* gbn1_m  = (const float*)d_in[20];
    const float* gbn1_v  = (const float*)d_in[21];
    const float* r2_wrel = (const float*)d_in[22];
    const float* r2_wroot= (const float*)d_in[23];
    const float* r2_b    = (const float*)d_in[24];
    const float* gbn2_g  = (const float*)d_in[25];
    const float* gbn2_b  = (const float*)d_in[26];
    const float* gbn2_m  = (const float*)d_in[27];
    const float* gbn2_v  = (const float*)d_in[28];
    const float* out_w   = (const float*)d_in[29];   // (512,1)
    const float* out_b   = (const float*)d_in[30];
    float* logits = (float*)d_out;

    // ---- workspace carve (256B aligned) ----
    char* base = (char*)d_ws;
    size_t off = 0;
    auto carve = [&](size_t bytes) -> char* {
        char* p = base + off;
        off += (bytes + 255) & ~(size_t)255;
        return p;
    };
    __bf16* wb1t   = (__bf16*)carve((size_t)HDIM * HDIM * 2);            // mlp_w1^T bf16 (256 x 256)
    __bf16* wb2t   = (__bf16*)carve((size_t)HDIM * HDIM * 2);            // mlp_w2^T bf16
    __bf16* wbr1t  = (__bf16*)carve((size_t)4 * HDIM * HDIM * 2);        // [wrel;wroot]^T layer1 (256 x 1024)
    __bf16* wbr2t  = (__bf16*)carve((size_t)4 * HDIM * HDIM * 2);        // layer2
    float*  cntR   = (float*) carve((size_t)NNODES * RREL * 4);          // row-agg inv counts (SCRE)
    float*  cntC   = (float*) carve((size_t)NNODES * RREL * 4);          // col-agg inv counts (RGCN 1&2)
    float*  relinv = (float*) carve((size_t)NNODES * 4);
    float*  agg    = (float*) carve((size_t)NNODES * RREL * HDIM * 4);   // 307MB; also SCRE scratch
    __bf16* bufA   = (__bf16*)carve((size_t)NNODES * HDIM * 2);          // z_mlp1 then z_gnn1
    __bf16* bufB   = (__bf16*)carve((size_t)NNODES * HDIM * 2);          // h

    const int TB = 256;
    dim3 gemmGrid((NNODES + 127) / 128, 4);
    int edgeWaveBlocks = (NEDGES + 7) / 8;          // 1 wave / edge, 8 waves / block
    long long aggN = (long long)NNODES * RREL * HDIM;

    // 1) weight transpose + bf16 conversion
    {
        long long t1 = (long long)HDIM * HDIM;       // 65536
        long long t4 = (long long)HDIM * 4 * HDIM;   // 262144
        transpose_w_kernel<<<(int)((t1 + TB - 1) / TB), TB, 0, stream>>>(mlp_w1, nullptr, wb1t, HDIM, HDIM);
        transpose_w_kernel<<<(int)((t1 + TB - 1) / TB), TB, 0, stream>>>(mlp_w2, nullptr, wb2t, HDIM, HDIM);
        transpose_w_kernel<<<(int)((t4 + TB - 1) / TB), TB, 0, stream>>>(r1_wrel, r1_wroot, wbr1t, RREL * HDIM, 4 * HDIM);
        transpose_w_kernel<<<(int)((t4 + TB - 1) / TB), TB, 0, stream>>>(r2_wrel, r2_wroot, wbr2t, RREL * HDIM, 4 * HDIM);
    }

    // 2) logits = out_b (head GEMMs accumulate into this)
    init_out_kernel<<<(NNODES + TB - 1) / TB, TB, 0, stream>>>(logits, out_b, NNODES);

    // 3) MLP branch: x -> bufA(bf16) -> head(out_w[0:256])
    gemm_bn_relu_kernel<float, HDIM, __bf16, 0, 0><<<gemmGrid, TB, 0, stream>>>(
        x, (const __bf16*)nullptr, wb1t, mlp_b1, mbn1_g, mbn1_b, mbn1_m, mbn1_v,
        bufA, nullptr, NNODES);
    gemm_bn_relu_kernel<__bf16, HDIM, __bf16, 0, 1><<<gemmGrid, TB, 0, stream>>>(
        bufA, (const __bf16*)nullptr, wb2t, mlp_b2, mbn2_g, mbn2_b, mbn2_m, mbn2_v,
        logits, out_w, NNODES);

    // 4) edge counts + inverses
    zero_f32_kernel<<<(NNODES * RREL + TB - 1) / TB, TB, 0, stream>>>(cntR, (long long)NNODES * RREL);
    zero_f32_kernel<<<(NNODES * RREL + TB - 1) / TB, TB, 0, stream>>>(cntC, (long long)NNODES * RREL);
    count_kernel<<<(NEDGES + TB - 1) / TB, TB, 0, stream>>>(ei, et, cntR, cntC, NEDGES);
    invert_scre_kernel<<<(NNODES + TB - 1) / TB, TB, 0, stream>>>(cntR, relinv, NNODES);
    invert_flat_kernel<<<(NNODES * RREL + TB - 1) / TB, TB, 0, stream>>>(cntC, NNODES * RREL);

    // 5) SCRE: tmp (= agg scratch) -> h = x - tmp*relinv  (bf16 in bufB)
    zero_f32_kernel<<<(int)((aggN + TB - 1) / TB), TB, 0, stream>>>(agg, aggN);
    scre_scatter_kernel<<<edgeWaveBlocks, TB, 0, stream>>>(ei, et, x, cntR, agg, NEDGES);
    compute_h_kernel<<<(NNODES * HDIM + TB - 1) / TB, TB, 0, stream>>>(
        x, agg, relinv, bufB, (long long)NNODES * HDIM);

    // 6) RGCN layer 1: agg = per-(col,rel) means of h[row]; K=1024 GEMM -> bufA(bf16)
    zero_f32_kernel<<<(int)((aggN + TB - 1) / TB), TB, 0, stream>>>(agg, aggN);
    rgcn_scatter_kernel<<<edgeWaveBlocks, TB, 0, stream>>>(ei, et, bufB, cntC, agg, NEDGES);
    gemm_bn_relu_kernel<float, RREL * HDIM, __bf16, HDIM, 0><<<gemmGrid, TB, 0, stream>>>(
        agg, bufB, wbr1t, r1_b, gbn1_g, gbn1_b, gbn1_m, gbn1_v, bufA, nullptr, NNODES);

    // 7) RGCN layer 2: same with z_gnn1 (bufA); head-fused epilogue with out_w[256:512]
    zero_f32_kernel<<<(int)((aggN + TB - 1) / TB), TB, 0, stream>>>(agg, aggN);
    rgcn_scatter_kernel<<<edgeWaveBlocks, TB, 0, stream>>>(ei, et, bufA, cntC, agg, NEDGES);
    gemm_bn_relu_kernel<float, RREL * HDIM, __bf16, HDIM, 1><<<gemmGrid, TB, 0, stream>>>(
        agg, bufA, wbr2t, r2_b, gbn2_g, gbn2_b, gbn2_m, gbn2_v, logits, out_w + HDIM, NNODES);
}
